// Encoder_61400852463725
// MI455X (gfx1250) — matrix operand order repaired and checked by reading, not compiled, not run on hardware
//
#include <hip/hip_runtime.h>

#define BN_INV 0.9999950000374997f

typedef __attribute__((ext_vector_type(16))) __bf16 v16bf;
typedef __attribute__((ext_vector_type(8)))  __bf16 v8bf;
typedef __attribute__((ext_vector_type(8)))  float  v8f;

// ---------------------------------------------------------------------------
// Stage 1: conv 3x3 (Cin=1) + BN + ReLU, direct (bandwidth bound, K=9)
// x: (8,1,256,256) -> e1: (8,32,256,256)
// ---------------------------------------------------------------------------
__global__ void __launch_bounds__(256) conv1_kernel(
    const float* __restrict__ x, const float* __restrict__ wgt,
    const float* __restrict__ g, const float* __restrict__ b,
    float* __restrict__ out)
{
    int idx = blockIdx.x * 256 + threadIdx.x;          // 16,777,216 total
    int w  = idx & 255;
    int h  = (idx >> 8) & 255;
    int co = (idx >> 16) & 31;
    int bI = idx >> 21;
    const float* xp = x + (size_t)bI * 65536;
    float acc = 0.f;
#pragma unroll
    for (int ky = 0; ky < 3; ++ky) {
        int ih = h + ky - 1;
        if ((unsigned)ih >= 256u) continue;
#pragma unroll
        for (int kx = 0; kx < 3; ++kx) {
            int iw = w + kx - 1;
            if ((unsigned)iw >= 256u) continue;
            acc += wgt[co * 9 + ky * 3 + kx] * xp[ih * 256 + iw];
        }
    }
    float y = acc * (g[co] * BN_INV) + b[co];
    out[idx] = fmaxf(y, 0.f);
}

// ---------------------------------------------------------------------------
// 2x2 maxpool, stride 2
// ---------------------------------------------------------------------------
__global__ void __launch_bounds__(256) maxpool_kernel(
    const float* __restrict__ in, float* __restrict__ out,
    int total, int Hin, int Win)
{
    int idx = blockIdx.x * 256 + threadIdx.x;
    if (idx >= total) return;
    int Ho = Hin >> 1, Wo = Win >> 1;
    int w  = idx % Wo;
    int h  = (idx / Wo) % Ho;
    int bc = idx / (Wo * Ho);
    const float* src = in + (size_t)bc * Hin * Win;
    int h2 = h * 2, w2 = w * 2;
    float m = src[h2 * Win + w2];
    m = fmaxf(m, src[h2 * Win + w2 + 1]);
    m = fmaxf(m, src[(h2 + 1) * Win + w2]);
    m = fmaxf(m, src[(h2 + 1) * Win + w2 + 1]);
    out[idx] = m;
}

// ---------------------------------------------------------------------------
// Implicit-GEMM conv 3x3 pad 1 + BN + ReLU via v_wmma_f32_16x16x32_bf16.
// GEMM: M=Cout, N=B*H*W, K=Cin*9 (K % 32 == 0 for all stages).
// Workgroup = 256 threads (8 waves). Block tile M=64 x N=128, K step 32.
// Each wave computes a 32x32 output tile = 4 WMMAs / K-chunk (2 A x 2 B frags).
// LDS: A tile [64][K=32] row-major, B tile [128][K=32] N-major so that each
// lane's fragment is two contiguous 16B-aligned 8-element runs (ds_load_b128).
// Fragment layouts per CDNA5 ISA 7.12.2 (wave32):
//   A 16x32 bf16 : row = lane&15 ; e<8 -> K=8*half+e ; e>=8 -> K=16+8*half+e-8
//   B 32x16 bf16 : col = lane&15 ; e<8 -> K=8*half+e ; e>=8 -> K=16+8*half+e-8
//   C/D 16x16 f32: col = lane&15 ; M = g + 8*half
// ---------------------------------------------------------------------------
__global__ void __launch_bounds__(256) conv_wmma_kernel(
    const float* __restrict__ in, const float* __restrict__ wgt,
    const float* __restrict__ gamma, const float* __restrict__ beta,
    float* __restrict__ out, int Cin, int Cout, int H, int W)
{
    const int K  = Cin * 9;
    const int HW = H * W;
    const int nBase  = blockIdx.x * 128;
    const int coBase = blockIdx.y * 64;

    __shared__ __align__(16) __bf16 As[64][40];    // [M][K], 80B row stride
    __shared__ __align__(16) __bf16 Bs[128][40];   // [N][K], 80B row stride

    const int tid  = threadIdx.x;
    const int lane = tid & 31;
    const int wave = tid >> 5;
    const int mw = wave & 1;       // M sub-tile of 32 (block M=64)
    const int nw = wave >> 1;      // N sub-tile of 32 (block N=128)
    const int half = lane >> 4;
    const int l15  = lane & 15;

    // ---- loop-invariant load indexing ------------------------------------
    // A tile: 64x32 elems, 8 per thread; kkA = tid&31 fixed, row varies.
    const int kkA   = tid & 31;
    const int rowA0 = tid >> 5;
    // B tile: 128x32 elems, 16 per thread; column fixed per thread.
    const int bcolT = tid & 127;
    const int kk0   = tid >> 7;                    // 0 or 1; kk = kk0 + 2*i
    const int nT    = nBase + bcolT;
    const int bIT   = nT / HW;
    const int remT  = nT - bIT * HW;
    const int hT    = remT / W;
    const int wT    = remT - hT * W;
    const float* inb = in + (size_t)bIT * Cin * HW;

    v8f acc[2][2];
#pragma unroll
    for (int f = 0; f < 2; ++f)
#pragma unroll
        for (int f2 = 0; f2 < 2; ++f2)
            acc[f][f2] = (v8f){0.f, 0.f, 0.f, 0.f, 0.f, 0.f, 0.f, 0.f};

    const int nChunks = K >> 5;
    for (int kc = 0; kc < nChunks; ++kc) {
        // --- stage A tile: weights contiguous in k = ci*9 + ky*3 + kx ---
#pragma unroll
        for (int i = 0; i < 8; ++i) {
            int row = rowA0 + 8 * i;
            As[row][kkA] = (__bf16)wgt[(size_t)(coBase + row) * K + kc * 32 + kkA];
        }
        // --- stage B tile: im2col gather with zero padding (N-major) ---
#pragma unroll
        for (int i = 0; i < 16; ++i) {
            int kk = kk0 + 2 * i;
            int kg = kc * 32 + kk;
            int ci = kg / 9;
            int r  = kg - ci * 9;
            int ky = r / 3;
            int kx = r - ky * 3;
            int ih = hT + ky - 1, iw = wT + kx - 1;
            float v = 0.f;
            if ((unsigned)ih < (unsigned)H && (unsigned)iw < (unsigned)W)
                v = inb[((size_t)ci * H + ih) * W + iw];
            Bs[bcolT][kk] = (__bf16)v;
        }
        if (kc + 1 < nChunks)   // gfx1250 global_prefetch_b8 on next weights
            __builtin_prefetch(&wgt[(size_t)(coBase + rowA0) * K + (kc + 1) * 32 + kkA], 0, 1);
        __syncthreads();

        // --- build fragments (two aligned 16B runs each) ---
        v16bf a[2], bfr[2];
#pragma unroll
        for (int f = 0; f < 2; ++f) {
            int arow = mw * 32 + f * 16 + l15;
            v8bf alo = *(const v8bf*)&As[arow][8 * half];
            v8bf ahi = *(const v8bf*)&As[arow][16 + 8 * half];
            a[f] = __builtin_shufflevector(alo, ahi,
                0, 1, 2, 3, 4, 5, 6, 7, 8, 9, 10, 11, 12, 13, 14, 15);
            int bcol = nw * 32 + f * 16 + l15;
            v8bf blo = *(const v8bf*)&Bs[bcol][8 * half];
            v8bf bhi = *(const v8bf*)&Bs[bcol][16 + 8 * half];
            bfr[f] = __builtin_shufflevector(blo, bhi,
                0, 1, 2, 3, 4, 5, 6, 7, 8, 9, 10, 11, 12, 13, 14, 15);
        }
        // --- 4 WMMAs: full 32x32 wave tile ---
#pragma unroll
        for (int f = 0; f < 2; ++f)
#pragma unroll
            for (int f2 = 0; f2 < 2; ++f2)
                acc[f][f2] = __builtin_amdgcn_wmma_f32_16x16x32_bf16(
                    false, a[f], false, bfr[f2], (short)0, acc[f][f2],
                    false, false);
        __syncthreads();
    }

    // --- epilogue: BN + ReLU, scatter D (col=lane&15, M=g+8*half) ---
#pragma unroll
    for (int f2 = 0; f2 < 2; ++f2) {
        int n   = nBase + nw * 32 + f2 * 16 + l15;
        int bI  = n / HW;
        int rem = n - bI * HW;
        int h   = rem / W;
        int w   = rem - h * W;
#pragma unroll
        for (int f = 0; f < 2; ++f) {
#pragma unroll
            for (int gi = 0; gi < 8; ++gi) {
                int co = coBase + mw * 32 + f * 16 + gi + 8 * half;
                float y = acc[f][f2][gi] * (gamma[co] * BN_INV) + beta[co];
                y = fmaxf(y, 0.f);
                out[((size_t)(bI * Cout + co) * H + h) * W + w] = y;
            }
        }
    }
}

// ---------------------------------------------------------------------------
// Quantum stage, part 1: per-channel real symmetric 16x16 matrix
//   M_c = sum_i zbar_i * (Re(U_c)_i^T Re(U_c)_i + Im(U_c)_i^T Im(U_c)_i)
// with U_c = P_TOT . kron(g0,g1,g2,g3), g_w = RZ(tz_w) RY(ty_w) H.
// One block (256 threads) per channel; U staged in LDS.
// ---------------------------------------------------------------------------
__device__ __forceinline__ int ptot(int b)
{
    int p = b;
    p = (p & 1) ? (p ^ 8) : p;   // CNOT(3,0)
    p = (p & 2) ? (p ^ 1) : p;   // CNOT(2,3)
    p = (p & 4) ? (p ^ 2) : p;   // CNOT(1,2)
    p = (p & 8) ? (p ^ 4) : p;   // CNOT(0,1)
    return p;
}

__global__ void __launch_bounds__(256) qmat_kernel(
    const float* __restrict__ qw0, const float* __restrict__ qw1,
    float* __restrict__ Mout)
{
    const int c = blockIdx.x;      // 256 channels
    const int t = threadIdx.x;
    __shared__ float ur[256], ui[256];   // U entry (i,j) at i*16+j

    const float ty[2] = { qw0[c * 2 + 0], qw0[c * 2 + 1] };
    const float tz[2] = { qw1[c * 2 + 0], qw1[c * 2 + 1] };
    const float is2 = 0.70710678118654752440f;

    float gre[4][2][2], gim[4][2][2];
#pragma unroll
    for (int w = 0; w < 4; ++w) {
        float tyw = ty[w & 1], tzw = tz[w & 1];
        float cc = cosf(0.5f * tyw), ss = sinf(0.5f * tyw);
        float aa = (cc - ss) * is2, bb = (cc + ss) * is2;
        float pc = cosf(0.5f * tzw), ps = sinf(0.5f * tzw);
        // row0 *= e^{-i tz/2}, row1 *= e^{+i tz/2};  RY*H = [[a,b],[b,-a]]
        gre[w][0][0] =  aa * pc;  gim[w][0][0] = -aa * ps;
        gre[w][0][1] =  bb * pc;  gim[w][0][1] = -bb * ps;
        gre[w][1][0] =  bb * pc;  gim[w][1][0] =  bb * ps;
        gre[w][1][1] = -aa * pc;  gim[w][1][1] = -aa * ps;
    }

    // thread t computes U[i][j], i = t>>4 (post-permutation row), j = t&15
    {
        int i = t >> 4, j = t & 15;
        int p = ptot(i);
        float re = 1.f, im = 0.f;
#pragma unroll
        for (int w = 0; w < 4; ++w) {
            int bi = (p >> (3 - w)) & 1;
            int bj = (j >> (3 - w)) & 1;
            float gr = gre[w][bi][bj], gi2 = gim[w][bi][bj];
            float nre = re * gr - im * gi2;
            float nim = re * gi2 + im * gr;
            re = nre; im = nim;
        }
        ur[t] = re; ui[t] = im;
    }
    __syncthreads();

    // thread t computes M[j][k], j = t>>4, k = t&15
    int j = t >> 4, k = t & 15;
    float acc = 0.f;
#pragma unroll
    for (int i = 0; i < 16; ++i) {
        float zb = 1.0f - 0.5f * (float)__popc(i);
        acc += zb * (ur[i * 16 + j] * ur[i * 16 + k] +
                     ui[i * 16 + j] * ui[i * 16 + k]);
    }
    Mout[c * 256 + t] = acc;
}

// ---------------------------------------------------------------------------
// Quantum stage, part 2: per pixel q = (p^T M p / max(||p||,1e-12)^2 + 1)/2,
// then BN+ReLU. One block per (b,c) plane (H=W=32); M and plane in LDS.
// ---------------------------------------------------------------------------
__global__ void __launch_bounds__(256) qeval_kernel(
    const float* __restrict__ e4, const float* __restrict__ Mmat,
    const float* __restrict__ g5, const float* __restrict__ b5,
    float* __restrict__ out)
{
    const int bc = blockIdx.x;        // b*256 + c
    const int c  = bc & 255;
    const int t  = threadIdx.x;
    __shared__ float Ms[256];
    __shared__ float plane[1024];

    Ms[t] = Mmat[c * 256 + t];
    const float* pl = e4 + (size_t)bc * 1024;
#pragma unroll
    for (int i = 0; i < 4; ++i) plane[t + i * 256] = pl[t + i * 256];
    __syncthreads();

    const float scale = g5[c] * BN_INV;
    const float bias  = b5[c];

    for (int rep = 0; rep < 4; ++rep) {
        int idx = t + rep * 256;
        int h = idx >> 5, w = idx & 31;
        float p[16];
        float nrm2 = 0.f;
#pragma unroll
        for (int di = 0; di < 4; ++di) {
#pragma unroll
            for (int dj = 0; dj < 4; ++dj) {
                int ih = h + di - 1, iw = w + dj - 1;   // pad ((1,2),(1,2))
                float v = 0.f;
                if ((unsigned)ih < 32u && (unsigned)iw < 32u)
                    v = plane[ih * 32 + iw];
                p[di * 4 + dj] = v;
                nrm2 += v * v;
            }
        }
        float den  = fmaxf(sqrtf(nrm2), 1e-12f);
        float inv2 = 1.0f / (den * den);
        float s = 0.f;
#pragma unroll
        for (int j = 0; j < 16; ++j) {
            float rd = 0.f;
#pragma unroll
            for (int k = 0; k < 16; ++k) rd += Ms[j * 16 + k] * p[k];
            s += p[j] * rd;
        }
        float q = 0.5f * (s * inv2 + 1.0f);
        out[(size_t)bc * 1024 + idx] = fmaxf(q * scale + bias, 0.f);
    }
}

// ---------------------------------------------------------------------------
// Host-side orchestration
// ---------------------------------------------------------------------------
extern "C" void kernel_launch(void* const* d_in, const int* in_sizes, int n_in,
                              void* d_out, int out_size, void* d_ws, size_t ws_size,
                              hipStream_t stream)
{
    const float* x   = (const float*)d_in[0];
    const float* w1  = (const float*)d_in[1];
    const float* g1  = (const float*)d_in[2];
    const float* b1  = (const float*)d_in[3];
    const float* w2  = (const float*)d_in[4];
    const float* g2  = (const float*)d_in[5];
    const float* b2  = (const float*)d_in[6];
    const float* w3  = (const float*)d_in[7];
    const float* g3  = (const float*)d_in[8];
    const float* b3  = (const float*)d_in[9];
    const float* w4  = (const float*)d_in[10];
    const float* g4  = (const float*)d_in[11];
    const float* b4  = (const float*)d_in[12];
    const float* qw0 = (const float*)d_in[13];
    const float* qw1 = (const float*)d_in[14];
    const float* g5  = (const float*)d_in[15];
    const float* b5  = (const float*)d_in[16];

    float* out  = (float*)d_out;
    float* e1   = out;                       // 8*32*256*256 = 16,777,216
    float* e2   = out + 16777216;            // 8*64*128*128 =  8,388,608
    float* e3   = out + 25165824;            // 8*128*64*64  =  4,194,304
    float* out4 = out + 29360128;            // 8*256*32*32  =  2,097,152

    float* ws = (float*)d_ws;
    float* p1 = ws;                          // 8*32*128*128 = 4,194,304
    float* p2 = p1 + 4194304;                // 8*64*64*64   = 2,097,152
    float* p3 = p2 + 2097152;                // 8*128*32*32  = 1,048,576
    float* e4 = p3 + 1048576;                // 8*256*32*32  = 2,097,152
    float* Mm = e4 + 2097152;                // 256*256      =    65,536

    // Stage 1
    conv1_kernel<<<65536, 256, 0, stream>>>(x, w1, g1, b1, e1);
    maxpool_kernel<<<16384, 256, 0, stream>>>(e1, p1, 4194304, 256, 256);

    // Stage 2: M=64, N=8*128*128=131072, K=288
    conv_wmma_kernel<<<dim3(1024, 1), 256, 0, stream>>>(
        p1, w2, g2, b2, e2, 32, 64, 128, 128);
    maxpool_kernel<<<8192, 256, 0, stream>>>(e2, p2, 2097152, 128, 128);

    // Stage 3: M=128, N=8*64*64=32768, K=576
    conv_wmma_kernel<<<dim3(256, 2), 256, 0, stream>>>(
        p2, w3, g3, b3, e3, 64, 128, 64, 64);
    maxpool_kernel<<<4096, 256, 0, stream>>>(e3, p3, 1048576, 64, 64);

    // Stage 4: M=256, N=8*32*32=8192, K=1152
    conv_wmma_kernel<<<dim3(64, 4), 256, 0, stream>>>(
        p3, w4, g4, b4, e4, 128, 256, 32, 32);

    // Quantum stage
    qmat_kernel<<<256, 256, 0, stream>>>(qw0, qw1, Mm);
    qeval_kernel<<<2048, 256, 0, stream>>>(e4, Mm, g5, b5, out4);
}
